// UncertaintyPrediction_48576080117830
// MI455X (gfx1250) — matrix operand order repaired and checked by reading, not compiled, not run on hardware
//
#include <hip/hip_runtime.h>
#include <hip/hip_bf16.h>

// ---------------- problem constants ----------------
#define BSZ   4
#define NSEQ  4096
#define CIN   384
#define CDSSP 64
#define CHID  128
#define NBLK  4
#define NTOK  (BSZ * NSEQ)        // 16384 tokens
#define NTILE (NTOK / 16)         // 1024 16-token tiles
#define WPB   4                   // waves per block (wave32)
#define MLP_BLOCKS (NTILE / WPB)  // 256 blocks of 128 threads

// LDS staging: 16 rows x 392 bf16 per wave (fits CIN=384 + pad for bank spread)
#define SSTG_IN 392
#define SSTG_H  136
#define SLAB    (16 * SSTG_IN)    // ushorts per wave slab

// workspace layout (ushort offsets for bf16 weights)
#define OFF_WIN   0
#define OFF_WINIT (OFF_WIN   + 128 * 384)
#define OFF_WDSSP (OFF_WINIT + 128 * 384)
#define OFF_W1    (OFF_WDSSP + 128 * 64)
#define OFF_W2    (OFF_W1    + 4 * 128 * 128)
#define OFF_WHEAD (OFF_W2    + 4 * 128 * 128)
#define WS_WEIGHT_BYTES ((OFF_WHEAD + 16 * 128) * 2)   // 479232 bytes
#define OFF_CORR_BYTES  WS_WEIGHT_BYTES                // f32 [NTOK][8] after weights

typedef __attribute__((ext_vector_type(16))) __bf16 v16bf;
typedef __attribute__((ext_vector_type(8)))  float  v8f;
typedef __attribute__((ext_vector_type(4)))  float  f32x4;   // native clang vector (NT-store OK)

union FragBF { uint4 u[2]; v16bf v; };

__device__ __forceinline__ unsigned short f2bf(float f) {
  unsigned int x = __float_as_uint(f);
  unsigned int r = x + 0x7fffu + ((x >> 16) & 1u);   // round-to-nearest-even
  return (unsigned short)(r >> 16);
}
__device__ __forceinline__ unsigned int pack2bf(float lo, float hi) {
  return (unsigned int)f2bf(lo) | ((unsigned int)f2bf(hi) << 16);
}
__device__ __forceinline__ float softplus_f(float x) {
  return fmaxf(x, 0.f) + log1pf(expf(-fabsf(x)));
}

// ---------------- kernel 1: zero the 256 MiB output (the real bottleneck) ----------------
__global__ void __launch_bounds__(256) zero_out_kernel(f32x4* __restrict__ out, long long nvec) {
  long long i = (long long)blockIdx.x * blockDim.x + threadIdx.x;
  const long long stride = (long long)gridDim.x * blockDim.x;
  f32x4 z = {0.f, 0.f, 0.f, 0.f};
  for (; i < nvec; i += stride) __builtin_nontemporal_store(z, &out[i]);
}

// ---------------- kernel 2: convert weights f32 -> bf16, pack heads ----------------
__global__ void __launch_bounds__(256) prep_weights_kernel(
    const float* __restrict__ W_in, const float* __restrict__ W_init,
    const float* __restrict__ W_dssp, const float* __restrict__ W1,
    const float* __restrict__ W2, const float* __restrict__ W_out,
    const float* __restrict__ W_mult, unsigned short* __restrict__ wb) {
  const int t = blockIdx.x * blockDim.x + threadIdx.x;
  const int stride = gridDim.x * blockDim.x;
  for (int i = t; i < 128 * 384; i += stride) wb[OFF_WIN + i]   = f2bf(W_in[i]);
  for (int i = t; i < 128 * 384; i += stride) wb[OFF_WINIT + i] = f2bf(W_init[i]);
  for (int i = t; i < 128 * 64;  i += stride) wb[OFF_WDSSP + i] = f2bf(W_dssp[i]);
  for (int i = t; i < 4 * 128 * 128; i += stride) wb[OFF_W1 + i] = f2bf(W1[i]);
  for (int i = t; i < 4 * 128 * 128; i += stride) wb[OFF_W2 + i] = f2bf(W2[i]);
  for (int i = t; i < 16 * 128; i += stride) {
    const int r = i >> 7, c = i & 127;
    float v = 0.f;
    if (r < 5)      v = W_out[r * 128 + c];
    else if (r < 7) v = W_mult[(r - 5) * 128 + c];
    wb[OFF_WHEAD + i] = f2bf(v);
  }
}

// ---------------- WMMA helpers ----------------
// A fragment (16x32 bf16, ISA layout): lane<16 row=lane holds K=[kb..kb+7] in v0..3,
// K=[kb+16..kb+23] in v4..7; lane>=16 holds K=[kb+8..kb+15], [kb+24..kb+31].
// B fragment (32x16 bf16): lane n (col) holds K=[kb..kb+7],[kb+16..kb+23] (lanes 0-15) /
// K=[kb+8..kb+15],[kb+24..kb+31] (lanes 16-31), same K order as A, from W[h][k].
template <int NCT>
__device__ __forceinline__ void gemm_acc(const unsigned short* __restrict__ Wb, int K,
                                         const unsigned short* __restrict__ sstage,
                                         int sstride, int lane, v8f* acc) {
  const int r = lane & 15, half = lane >> 4;
  for (int kb = 0; kb < K; kb += 32) {
    FragBF A;
    A.u[0] = *(const uint4*)(sstage + r * sstride + kb + half * 8);
    A.u[1] = *(const uint4*)(sstage + r * sstride + kb + 16 + half * 8);
#pragma unroll
    for (int ct = 0; ct < NCT; ++ct) {
      FragBF B;
      const unsigned short* wrow = Wb + (ct * 16 + r) * K + kb + half * 8;
      B.u[0] = *(const uint4*)(wrow);
      B.u[1] = *(const uint4*)(wrow + 16);
      acc[ct] = __builtin_amdgcn_wmma_f32_16x16x32_bf16(
          /*neg_a=*/false, A.v, /*neg_b=*/false, B.v,
          /*c_mod=*/(short)0, acc[ct], /*reuse_a=*/false, /*reuse_b=*/false);
    }
  }
}

__device__ __forceinline__ void stage_input_relu(const f32x4* __restrict__ src4, int row_vecs,
                                                 unsigned short* __restrict__ stg,
                                                 int sstride, int lane) {
  for (int r = 0; r < 16; ++r)
    for (int c = lane; c < row_vecs; c += 32) {
      f32x4 v = src4[r * row_vecs + c];
      v.x = fmaxf(v.x, 0.f); v.y = fmaxf(v.y, 0.f);
      v.z = fmaxf(v.z, 0.f); v.w = fmaxf(v.w, 0.f);
      uint2 p; p.x = pack2bf(v.x, v.y); p.y = pack2bf(v.z, v.w);
      *(uint2*)(stg + r * sstride + c * 4) = p;
    }
}

// C/D layout: lane L, VGPR v -> (row M = v + 8*(L>>4), col N = L&15)
__device__ __forceinline__ void stage_acc_relu(const v8f* acc, unsigned short* __restrict__ stg,
                                               int sstride, int lane) {
  const int col = lane & 15, half = lane >> 4;
#pragma unroll
  for (int ct = 0; ct < 8; ++ct)
#pragma unroll
    for (int v = 0; v < 8; ++v) {
      const int row = v + 8 * half;
      stg[row * sstride + ct * 16 + col] = f2bf(fmaxf(acc[ct][v], 0.f));
    }
}

// ---------------- kernel 3: fused MLP, one wave per 16-token tile ----------------
__global__ void __launch_bounds__(WPB * 32) mlp_kernel(
    const float* __restrict__ s, const float* __restrict__ s_initial,
    const float* __restrict__ dssp, const unsigned short* __restrict__ wb,
    const float* __restrict__ b_in, const float* __restrict__ b_init,
    const float* __restrict__ b_dssp, const float* __restrict__ b1,
    const float* __restrict__ b2, const float* __restrict__ b_out,
    const float* __restrict__ b_mult, float* __restrict__ corr_ws) {
  __shared__ unsigned short lds[WPB * SLAB];
  const int wave = threadIdx.x >> 5, lane = threadIdx.x & 31;
  const int col = lane & 15, half = lane >> 4;
  unsigned short* stg = lds + wave * SLAB;
  const int tile = blockIdx.x * WPB + wave;
  const int tok0 = tile * 16;

  v8f h[8];
#pragma unroll
  for (int i = 0; i < 8; ++i) h[i] = {};

  // --- input projections: h = Win*relu(s) + Winit*relu(s0) + Wdssp*relu(dssp) ---
  stage_input_relu((const f32x4*)s + tok0 * (CIN / 4), CIN / 4, stg, SSTG_IN, lane);
  gemm_acc<8>(wb + OFF_WIN, CIN, stg, SSTG_IN, lane, h);
  stage_input_relu((const f32x4*)s_initial + tok0 * (CIN / 4), CIN / 4, stg, SSTG_IN, lane);
  gemm_acc<8>(wb + OFF_WINIT, CIN, stg, SSTG_IN, lane, h);
  stage_input_relu((const f32x4*)dssp + tok0 * (CDSSP / 4), CDSSP / 4, stg, SSTG_IN, lane);
  gemm_acc<8>(wb + OFF_WDSSP, CDSSP, stg, SSTG_IN, lane, h);
#pragma unroll
  for (int ct = 0; ct < 8; ++ct) {
    const float b = b_in[ct * 16 + col] + b_init[ct * 16 + col] + b_dssp[ct * 16 + col];
#pragma unroll
    for (int v = 0; v < 8; ++v) h[ct][v] += b;
  }

  // --- resnet blocks: h += W2*relu(W1*relu(h)+b1)+b2, W2 GEMM accumulates into h directly ---
  for (int blk = 0; blk < NBLK; ++blk) {
    stage_acc_relu(h, stg, SSTG_H, lane);
    v8f t[8];
#pragma unroll
    for (int i = 0; i < 8; ++i) t[i] = {};
    gemm_acc<8>(wb + OFF_W1 + blk * 128 * 128, CHID, stg, SSTG_H, lane, t);
#pragma unroll
    for (int ct = 0; ct < 8; ++ct) {
      const float b = b1[blk * 128 + ct * 16 + col];
#pragma unroll
      for (int v = 0; v < 8; ++v) t[ct][v] += b;
    }
    stage_acc_relu(t, stg, SSTG_H, lane);
    gemm_acc<8>(wb + OFF_W2 + blk * 128 * 128, CHID, stg, SSTG_H, lane, h);
#pragma unroll
    for (int ct = 0; ct < 8; ++ct) {
      const float b = b2[blk * 128 + ct * 16 + col];
#pragma unroll
      for (int v = 0; v < 8; ++v) h[ct][v] += b;
    }
  }

  // --- heads: packed [W_out;W_mult] 16x128, one WMMA column tile ---
  stage_acc_relu(h, stg, SSTG_H, lane);
  v8f ho = {};
  gemm_acc<1>(wb + OFF_WHEAD, CHID, stg, SSTG_H, lane, &ho);
  float bh = 0.f;
  if (col < 5) bh = b_out[col];
  else if (col < 7) bh = b_mult[col - 5];
#pragma unroll
  for (int v = 0; v < 8; ++v) ho[v] += bh;
  if (col < 7) {
#pragma unroll
    for (int v = 0; v < 8; ++v) {
      const int row = v + 8 * half;
      corr_ws[(tok0 + row) * 8 + col] = ho[v];
    }
  }
}

// ---------------- kernel 4: softplus + banded scatter ----------------
__global__ void __launch_bounds__(256) scatter_kernel(const float* __restrict__ corr_ws,
                                                      const float* __restrict__ sm_p,
                                                      float* __restrict__ out) {
  const int tid = blockIdx.x * blockDim.x + threadIdx.x;
  if (tid >= NTOK) return;
  const int b = tid >> 12, i = tid & (NSEQ - 1);
  const float sm = sm_p[0];
  const float* c = corr_ws + tid * 8;
  const float d0 = softplus_f(c[0]) * softplus_f(c[5]) + softplus_f(c[6]);
  const size_t rowbase = ((size_t)b * NSEQ + i) * NSEQ;
  out[rowbase + i] = sm * d0;
  if (i >= 1) out[rowbase + i - 1] = sm * (c[1] + corr_ws[(size_t)(tid - 1) * 8 + 3]);
  if (i >= 2) out[rowbase + i - 2] = sm * (c[2] + corr_ws[(size_t)(tid - 2) * 8 + 4]);
}

// ---------------- launch ----------------
extern "C" void kernel_launch(void* const* d_in, const int* in_sizes, int n_in,
                              void* d_out, int out_size, void* d_ws, size_t ws_size,
                              hipStream_t stream) {
  const float* s       = (const float*)d_in[0];
  const float* s_init  = (const float*)d_in[1];
  const float* dssp    = (const float*)d_in[2];
  const float* W_in    = (const float*)d_in[3];
  const float* b_in    = (const float*)d_in[4];
  const float* W_init  = (const float*)d_in[5];
  const float* b_init  = (const float*)d_in[6];
  const float* W_dssp  = (const float*)d_in[7];
  const float* b_dssp  = (const float*)d_in[8];
  const float* W1      = (const float*)d_in[9];
  const float* b1      = (const float*)d_in[10];
  const float* W2      = (const float*)d_in[11];
  const float* b2      = (const float*)d_in[12];
  const float* W_out   = (const float*)d_in[13];
  const float* b_out   = (const float*)d_in[14];
  const float* W_mult  = (const float*)d_in[15];
  const float* b_mult  = (const float*)d_in[16];
  const float* sm      = (const float*)d_in[17];

  unsigned short* wb = (unsigned short*)d_ws;
  float* corr_ws = (float*)((char*)d_ws + OFF_CORR_BYTES);

  const long long nvec = (long long)out_size / 4;   // float4 count (256 MiB / 16)
  zero_out_kernel<<<4096, 256, 0, stream>>>((f32x4*)d_out, nvec);
  prep_weights_kernel<<<128, 256, 0, stream>>>(W_in, W_init, W_dssp, W1, W2, W_out, W_mult, wb);
  mlp_kernel<<<MLP_BLOCKS, WPB * 32, 0, stream>>>(s, s_init, dssp, wb, b_in, b_init, b_dssp,
                                                  b1, b2, b_out, b_mult, corr_ws);
  scatter_kernel<<<NTOK / 256, 256, 0, stream>>>(corr_ws, sm, (float*)d_out);
}
